// MultiScaleRetention_37400575213788
// MI455X (gfx1250) — compile-verified, matrix operand order checked by reading, not run on hardware
//
#include <hip/hip_runtime.h>
#include <cstdint>
#include <cstddef>

typedef __bf16 bf16;
typedef __attribute__((ext_vector_type(16))) __bf16 v16bf;
typedef __attribute__((ext_vector_type(8)))  __bf16 v8bf;
typedef __attribute__((ext_vector_type(8)))  float  v8f;
typedef __attribute__((ext_vector_type(4)))  unsigned int u32x4;
typedef __attribute__((ext_vector_type(8)))  int i32x8;
typedef __attribute__((ext_vector_type(4)))  int i32x4;
typedef unsigned long long u64;

union Frag {
    v16bf v;
    v8bf  h[2];
};

// problem sizes
#define BATCH   4
#define SEQ     2048
#define DMODEL  2048
#define NHEAD   16
#define HEADDIM 128
#define MROWS   (BATCH * SEQ)          // 8192
#define NCHUNK  16                     // scan chunks
#define CHLEN   (SEQ / NCHUNK)         // 128

// GEMM tiling: block 128x256x64, 8 waves, wave tile 32x128 (2x8 WMMA)
#define BM 128
#define BN 256
#define BK 64
#define LDSTR 72                       // LDS row stride in bf16 (144B, 16B aligned)
#define A_ELEMS (BM * LDSTR)           //  9216 elems = 18432 B
#define B_ELEMS (BN * LDSTR)           // 18432 elems = 36864 B
#define STG_ELEMS (A_ELEMS + B_ELEMS)  // 27648 elems = 55296 B per buffer

#if defined(__AMDGCN__) && __has_builtin(__builtin_amdgcn_tensor_load_to_lds) && \
    __has_builtin(__builtin_amdgcn_s_wait_tensorcnt)
#define TDM_PATH 1
#else
#define TDM_PATH 0
#endif

template<typename T> __device__ __forceinline__ T cvt_out(float f);
template<> __device__ __forceinline__ float cvt_out<float>(float f) { return f; }
template<> __device__ __forceinline__ bf16  cvt_out<bf16 >(float f) { return (bf16)f; }

// ---------------------------------------------------------------------------
// fp32 -> bf16 cast
// ---------------------------------------------------------------------------
__global__ void cast_f32_bf16_kernel(const float* __restrict__ in,
                                     bf16* __restrict__ out, int n) {
    int i = blockIdx.x * blockDim.x + threadIdx.x;
    if (i < n) out[i] = (bf16)in[i];
}

#if TDM_PATH
// ---------------------------------------------------------------------------
// Tensor Data Mover: load a 2D tile (tile_dim0=64 k-elems x tile_rows, bf16)
// from a K-strided tensor into LDS at lds_byte_off.
// LDS padding: after every 32 DWORDs (one 128B row) insert 4 DWORDs (16B)
// -> LDS row stride = 144B = 72 bf16 elems, 16B aligned, 36-bank skew.
// D# layout per CDNA5 ISA 8.3-8.6 (groups 2/3 zero: 2-D tensor).
// ---------------------------------------------------------------------------
__device__ __forceinline__ void tdm_load_tile(u64 gaddr, unsigned lds_byte_off,
                                              unsigned Kdim, unsigned rows,
                                              unsigned tile_rows) {
    u32x4 g0;
    g0[0] = 1u;                                           // count=1, user mode
    g0[1] = lds_byte_off;                                 // lds_addr
    g0[2] = (unsigned)(gaddr & 0xFFFFFFFFull);            // global_addr[31:0]
    g0[3] = (unsigned)((gaddr >> 32) & 0x01FFFFFFull)     // global_addr[56:32]
          | (2u << 30);                                   // type=2 ("image")
    i32x8 g1;
    g1[0] = (int)((1u << 16)      // data_size = 2 bytes
                | (1u << 20)      // pad_enable
                | (4u << 22)      // pad_interval: 32 DWORDs (one row)
                | (3u << 25));    // pad_amount: 4 DWORDs (16B)
    g1[1] = (int)((Kdim & 0xFFFFu) << 16);                  // tensor_dim0[15:0]
    g1[2] = (int)((Kdim >> 16) | ((rows & 0xFFFFu) << 16)); // dim0 hi | dim1 lo
    g1[3] = (int)((rows >> 16) | (64u << 16));              // dim1 hi | tile_dim0=64
    g1[4] = (int)tile_rows;                                 // tile_dim1 (tile_dim2=0)
    g1[5] = (int)Kdim;                                      // tensor_dim0_stride lo32
    g1[6] = 0;                                              // stride hi | dim1_stride lo
    g1[7] = 0;
    i32x4 z4 = {0, 0, 0, 0};
#if defined(__clang_major__) && __clang_major__ >= 23
    i32x8 z8 = {0, 0, 0, 0, 0, 0, 0, 0};
    __builtin_amdgcn_tensor_load_to_lds(g0, g1, z4, z4, z8, 0);
#else
    __builtin_amdgcn_tensor_load_to_lds(g0, g1, z4, z4, 0);
#endif
}
#endif

// ---------------------------------------------------------------------------
// one staged K=64 slab: 2 sub-steps of K=32, 2x8 WMMA tiles per wave
// ---------------------------------------------------------------------------
__device__ __forceinline__ void compute_stage(const bf16* As, const bf16* Bs,
                                              v8f (&acc)[2][8],
                                              int wm, int wn, int r, int g) {
#pragma unroll
    for (int ks = 0; ks < BK; ks += 32) {
        Frag fa[2], fb[8];
#pragma unroll
        for (int m = 0; m < 2; ++m) {
            const int row = wm * 32 + m * 16 + r;
            fa[m].h[0] = *(const v8bf*)&As[row * LDSTR + ks + g * 8];
            fa[m].h[1] = *(const v8bf*)&As[row * LDSTR + ks + 16 + g * 8];
        }
#pragma unroll
        for (int n = 0; n < 8; ++n) {
            const int wrow = wn * 128 + n * 16 + r;       // W row == output col
            fb[n].h[0] = *(const v8bf*)&Bs[wrow * LDSTR + ks + g * 8];
            fb[n].h[1] = *(const v8bf*)&Bs[wrow * LDSTR + ks + 16 + g * 8];
        }
#pragma unroll
        for (int m = 0; m < 2; ++m)
#pragma unroll
            for (int n = 0; n < 8; ++n)
                acc[m][n] = __builtin_amdgcn_wmma_f32_16x16x32_bf16(
                    false, fa[m].v, false, fb[n].v,
                    (short)0, acc[m][n], false, false);
    }
}

// ---------------------------------------------------------------------------
// C[M,N] = A[M,K] @ W[N,K]^T   (bf16 in, f32 accumulate via WMMA, OutT out)
// ---------------------------------------------------------------------------
template<typename OutT>
__global__ __launch_bounds__(256)
void gemm_bf16_kernel(const bf16* __restrict__ A,
                      const bf16* __restrict__ Bw,
                      OutT* __restrict__ C,
                      int M, int N, int K) {
    const int tid  = threadIdx.x;
    const int lane = tid & 31;
    const int wid  = tid >> 5;
    const int wm   = wid & 3;        // wave row  (4 waves along M)
    const int wn   = wid >> 2;       // wave col  (2 waves along N)
    const int r    = lane & 15;
    const int g    = lane >> 4;

    const int m0 = blockIdx.y * BM;
    const int n0 = blockIdx.x * BN;

    // [A0|B0|A1|B1]: per buffer A=18432B, B=36864B; total 108KB
    __shared__ bf16 smem[2 * STG_ELEMS];

    v8f acc[2][8] = {};

#if TDM_PATH
    const int ktiles = K / BK;

    if (wid == 0) {
        tdm_load_tile((u64)(uintptr_t)(A  + (size_t)m0 * K), 0u,
                      (unsigned)K, (unsigned)M, BM);
        tdm_load_tile((u64)(uintptr_t)(Bw + (size_t)n0 * K), A_ELEMS * 2u,
                      (unsigned)K, (unsigned)N, BN);
    }

    int buf = 0;
    for (int kt = 0; kt < ktiles; ++kt) {
        if (wid == 0) {
            if (kt + 1 < ktiles) {
                const int kk = (kt + 1) * BK;
                const unsigned off = (buf ^ 1) ? (unsigned)(STG_ELEMS * 2) : 0u;
                tdm_load_tile((u64)(uintptr_t)(A  + (size_t)m0 * K + kk), off,
                              (unsigned)K, (unsigned)M, BM);
                tdm_load_tile((u64)(uintptr_t)(Bw + (size_t)n0 * K + kk),
                              off + A_ELEMS * 2u, (unsigned)K, (unsigned)N, BN);
                __builtin_amdgcn_s_wait_tensorcnt(2);     // current pair landed
            } else {
                __builtin_amdgcn_s_wait_tensorcnt(0);
            }
        }
        __syncthreads();                                  // publish tiles

        const bf16* As = smem + buf * STG_ELEMS;
        compute_stage(As, As + A_ELEMS, acc, wm, wn, r, g);

        __syncthreads();                                  // done reading buf
        buf ^= 1;
    }
#else
    // fallback: thread-cooperative staging, single buffer
    bf16* As = smem;
    bf16* Bs = smem + A_ELEMS;
    for (int kk = 0; kk < K; kk += BK) {
        __syncthreads();
        for (int c = tid; c < (BM * BK) / 8; c += 256) {
            const int row = c >> 3, seg = c & 7;
            *(v8bf*)&As[row * LDSTR + seg * 8] =
                *(const v8bf*)(A + (size_t)(m0 + row) * K + kk + seg * 8);
        }
        for (int c = tid; c < (BN * BK) / 8; c += 256) {
            const int row = c >> 3, seg = c & 7;
            *(v8bf*)&Bs[row * LDSTR + seg * 8] =
                *(const v8bf*)(Bw + (size_t)(n0 + row) * K + kk + seg * 8);
        }
        __syncthreads();
        compute_stage(As, Bs, acc, wm, wn, r, g);
    }
#endif

    // epilogue: C/D layout: VGPR j -> M = g*8 + j, N = r
#pragma unroll
    for (int m = 0; m < 2; ++m)
#pragma unroll
        for (int n = 0; n < 8; ++n)
#pragma unroll
            for (int j = 0; j < 8; ++j) {
                const int row = m0 + wm * 32 + m * 16 + g * 8 + j;
                const int col = n0 + wn * 128 + n * 16 + r;
                C[(size_t)row * N + col] = cvt_out<OutT>(acc[m][n][j]);
            }
}

// ---------------------------------------------------------------------------
// decay scan, pass A: per-(b,h,chunk) local scan with zero init, in place
// ---------------------------------------------------------------------------
__global__ __launch_bounds__(128)
void scanA_kernel(bf16* __restrict__ v, const float* __restrict__ beta) {
    const int j     = threadIdx.x;
    const int chunk = blockIdx.x & (NCHUNK - 1);
    const int h     = (blockIdx.x >> 4) & (NHEAD - 1);
    const int b     = blockIdx.x >> 8;
    const float lam = 1.0f / (1.0f + __expf(-beta[h]));
    float s = 0.0f;
    size_t base = ((size_t)(b * SEQ + chunk * CHLEN)) * DMODEL + h * HEADDIM + j;
    for (int t = 0; t < CHLEN; ++t) {
        const size_t idx = base + (size_t)t * DMODEL;
        s = lam * s + (float)v[idx];
        v[idx] = (bf16)s;        // local state (carry applied in pass C)
    }
}

// ---------------------------------------------------------------------------
// decay scan, pass B: propagate chunk carries (16 sequential chunks)
// ---------------------------------------------------------------------------
__global__ __launch_bounds__(128)
void scanB_kernel(const bf16* __restrict__ v, const float* __restrict__ beta,
                  float* __restrict__ carr) {
    const int j = threadIdx.x;
    const int h = blockIdx.x & (NHEAD - 1);
    const int b = blockIdx.x >> 4;
    const float lam = 1.0f / (1.0f + __expf(-beta[h]));
    float lam128 = lam;
#pragma unroll
    for (int i = 0; i < 7; ++i) lam128 *= lam128;   // lam^128
    float c = 0.0f;
    for (int ch = 0; ch < NCHUNK; ++ch) {
        carr[((size_t)blockIdx.x * NCHUNK + ch) * HEADDIM + j] = c;
        const size_t idx = ((size_t)(b * SEQ + ch * CHLEN + CHLEN - 1)) * DMODEL
                         + h * HEADDIM + j;
        c = lam128 * c + (float)v[idx];   // state at end of chunk ch
    }
}

// ---------------------------------------------------------------------------
// decay scan, pass C: apply carries, y_raw = q * state (in place over v)
// ---------------------------------------------------------------------------
__global__ __launch_bounds__(128)
void scanC_kernel(bf16* __restrict__ v, const bf16* __restrict__ q,
                  const float* __restrict__ beta, const float* __restrict__ carr) {
    const int j     = threadIdx.x;
    const int chunk = blockIdx.x & (NCHUNK - 1);
    const int h     = (blockIdx.x >> 4) & (NHEAD - 1);
    const int b     = blockIdx.x >> 8;
    const float lam = 1.0f / (1.0f + __expf(-beta[h]));
    const float c   = carr[(((size_t)(b * NHEAD + h)) * NCHUNK + chunk) * HEADDIM + j];
    float p = lam;
    size_t base = ((size_t)(b * SEQ + chunk * CHLEN)) * DMODEL + h * HEADDIM + j;
    for (int t = 0; t < CHLEN; ++t) {
        const size_t idx = base + (size_t)t * DMODEL;
        const float s = (float)v[idx] + p * c;   // true state
        p *= lam;
        v[idx] = (bf16)((float)q[idx] * s);      // y_raw
    }
}

// ---------------------------------------------------------------------------
// LayerNorm over D + SiLU gate:  out = LN(y_raw) * silu(g)   (bf16 out)
// ---------------------------------------------------------------------------
__global__ __launch_bounds__(256)
void lngate_kernel(const bf16* __restrict__ y, const bf16* __restrict__ gpre,
                   const float* __restrict__ lnw, const float* __restrict__ lnb,
                   bf16* __restrict__ out) {
    const int row = blockIdx.x;
    const int tid = threadIdx.x;
    const int lane = tid & 31, wid = tid >> 5;
    const size_t base = (size_t)row * DMODEL;

    float vals[8];
    float s = 0.0f, s2 = 0.0f;
#pragma unroll
    for (int i = 0; i < 8; ++i) {
        const int col = tid + i * 256;
        const float f = (float)y[base + col];
        vals[i] = f;
        s += f; s2 += f * f;
    }
#pragma unroll
    for (int off = 16; off > 0; off >>= 1) {
        s  += __shfl_xor(s,  off, 32);
        s2 += __shfl_xor(s2, off, 32);
    }
    __shared__ float red[16];
    if (lane == 0) { red[wid] = s; red[8 + wid] = s2; }
    __syncthreads();
    float ts = 0.0f, ts2 = 0.0f;
#pragma unroll
    for (int w = 0; w < 8; ++w) { ts += red[w]; ts2 += red[8 + w]; }
    const float mu  = ts * (1.0f / DMODEL);
    const float var = ts2 * (1.0f / DMODEL) - mu * mu;
    const float inv = rsqrtf(var + 1e-5f);
#pragma unroll
    for (int i = 0; i < 8; ++i) {
        const int col = tid + i * 256;
        const float gv   = (float)gpre[base + col];
        const float gate = gv / (1.0f + __expf(-gv));     // silu
        const float yn   = (vals[i] - mu) * inv * lnw[col] + lnb[col];
        out[base + col] = (bf16)(yn * gate);
    }
}

// ---------------------------------------------------------------------------
extern "C" void kernel_launch(void* const* d_in, const int* in_sizes, int n_in,
                              void* d_out, int out_size, void* d_ws, size_t ws_size,
                              hipStream_t stream) {
    (void)in_sizes; (void)n_in; (void)out_size; (void)ws_size;

    const float* x    = (const float*)d_in[0];
    const float* Wq   = (const float*)d_in[1];
    const float* Wv   = (const float*)d_in[2];
    const float* Wg   = (const float*)d_in[3];
    const float* Wo   = (const float*)d_in[4];
    const float* beta = (const float*)d_in[5];
    const float* lnw  = (const float*)d_in[6];
    const float* lnb  = (const float*)d_in[7];
    float* out = (float*)d_out;

    const size_t EX = (size_t)MROWS * DMODEL;    // 16,777,216
    const size_t EW = (size_t)DMODEL * DMODEL;   //  4,194,304

    bf16* xb   = (bf16*)d_ws;
    bf16* wb   = xb + EX;                        // 4 weights, bf16
    bf16* qb   = wb + 4 * EW;
    bf16* vb   = qb + EX;
    bf16* gb   = vb + EX;
    float* carr = (float*)(gb + EX);             // [B*H, 16, 128] fp32

    // 1) casts
    {
        const int T = 256;
        cast_f32_bf16_kernel<<<(int)((EX + T - 1) / T), T, 0, stream>>>(x, xb, (int)EX);
        cast_f32_bf16_kernel<<<(int)((EW + T - 1) / T), T, 0, stream>>>(Wq, wb + 0 * EW, (int)EW);
        cast_f32_bf16_kernel<<<(int)((EW + T - 1) / T), T, 0, stream>>>(Wv, wb + 1 * EW, (int)EW);
        cast_f32_bf16_kernel<<<(int)((EW + T - 1) / T), T, 0, stream>>>(Wg, wb + 2 * EW, (int)EW);
        cast_f32_bf16_kernel<<<(int)((EW + T - 1) / T), T, 0, stream>>>(Wo, wb + 3 * EW, (int)EW);
    }

    // 2) projections: q, v, g  (bf16 out)
    {
        dim3 grid(DMODEL / BN, MROWS / BM);   // 8 x 64
        gemm_bf16_kernel<bf16><<<grid, 256, 0, stream>>>(xb, wb + 0 * EW, qb, MROWS, DMODEL, DMODEL);
        gemm_bf16_kernel<bf16><<<grid, 256, 0, stream>>>(xb, wb + 1 * EW, vb, MROWS, DMODEL, DMODEL);
        gemm_bf16_kernel<bf16><<<grid, 256, 0, stream>>>(xb, wb + 2 * EW, gb, MROWS, DMODEL, DMODEL);
    }

    // 3) chunk-parallel decay scan + q*state  (vb -> y_raw, in place)
    scanA_kernel<<<BATCH * NHEAD * NCHUNK, 128, 0, stream>>>(vb, beta);
    scanB_kernel<<<BATCH * NHEAD, 128, 0, stream>>>(vb, beta, carr);
    scanC_kernel<<<BATCH * NHEAD * NCHUNK, 128, 0, stream>>>(vb, qb, beta, carr);

    // 4) LayerNorm + SiLU gate  (reuse qb as gated-y buffer)
    lngate_kernel<<<MROWS, 256, 0, stream>>>(vb, gb, lnw, lnb, qb);

    // 5) output projection: out = y @ Wo^T  (fp32 out)
    {
        dim3 grid(DMODEL / BN, MROWS / BM);
        gemm_bf16_kernel<float><<<grid, 256, 0, stream>>>(qb, wb + 3 * EW, out, MROWS, DMODEL, DMODEL);
    }
}
